// SparseAttention_52407190946375
// MI455X (gfx1250) — matrix operand order repaired and checked by reading, not compiled, not run on hardware
//
#include <hip/hip_runtime.h>

typedef __attribute__((ext_vector_type(16))) _Float16 v16h;
typedef __attribute__((ext_vector_type(8)))  _Float16 v8h;
typedef __attribute__((ext_vector_type(8)))  float    v8f;

#define DM    1024            // d_model
#define NH    16              // heads
#define HD    64              // head dim
#define BB    2               // batch
#define TT    2048            // tgt len
#define SS    2048            // src len
#define NG    32              // global tokens
#define SPN   (NG + SS)       // 2080 concat source positions
#define VT_LD 2336            // 64 guard + 2080 + 192 guard (mult of 8)
#define VT_OFF 64
#define MROWS (BB * TT)       // 4096

static __device__ __forceinline__ v8h ld8(const _Float16* p) { return *(const v8h*)p; }
static __device__ __forceinline__ v16h cmb(v8h a, v8h b) {
  return __builtin_shufflevector(a, b, 0,1,2,3,4,5,6,7,8,9,10,11,12,13,14,15);
}
static __device__ __forceinline__ v8f wmma16(v16h a, v16h b, v8f c) {
  // D = A x B + C, fp32 accum, f16 inputs
  return __builtin_amdgcn_wmma_f32_16x16x32_f16(false, a, false, b, (short)0, c, false, false);
}
static __device__ __forceinline__ unsigned pkh2(float a, float b) {
  union { _Float16 h[2]; unsigned u; } cv;
  cv.h[0] = (_Float16)a; cv.h[1] = (_Float16)b;
  return cv.u;
}

// ---------------- elementwise helpers ----------------
__global__ void cast_f32_f16(const float* __restrict__ in, _Float16* __restrict__ out, int n) {
  int i = blockIdx.x * blockDim.x + threadIdx.x;
  if (i < n) out[i] = (_Float16)in[i];
}

__global__ void zero_f16(_Float16* __restrict__ p, int n) {
  int i = blockIdx.x * blockDim.x + threadIdx.x;
  if (i < n) p[i] = (_Float16)0.f;
}

// rows 0..31 of concatenated K get the raw global tokens
__global__ void kcat_globals(const _Float16* __restrict__ g, _Float16* __restrict__ Kc) {
  int i = blockIdx.x * blockDim.x + threadIdx.x;
  if (i >= BB * NG * DM) return;
  int b = i / (NG * DM);
  int r = i - b * (NG * DM);
  Kc[(size_t)b * SPN * DM + r] = g[r];
}

// V^T[b][h][d][s'=0..31] = global_tokens[s'][h*64+d]
__global__ void vt_globals(const _Float16* __restrict__ g, _Float16* __restrict__ Vt) {
  int i = blockIdx.x * blockDim.x + threadIdx.x;
  if (i >= BB * NH * HD * NG) return;
  int s = i & 31, d = (i >> 5) & 63, h = (i >> 11) & 15, b = i >> 15;
  Vt[(size_t)((b * NH + h) * HD + d) * VT_LD + VT_OFF + s] = g[s * DM + h * HD + d];
}

// gather strided columns: Vs[b][h][d][j] = V^T[b][h][d][32 + 64*j]
__global__ void vts_fill(const _Float16* __restrict__ Vt, _Float16* __restrict__ Vs) {
  int i = blockIdx.x * blockDim.x + threadIdx.x;
  if (i >= BB * NH * HD * NG) return;
  int j = i & 31, d = (i >> 5) & 63, h = (i >> 11) & 15, b = i >> 15;
  Vs[i] = Vt[(size_t)((b * NH + h) * HD + d) * VT_LD + VT_OFF + NG + 64 * j];
}

// ---------------- WMMA GEMM: C[M,N] = A[M,K] * W[N,K]^T + bias ----------------
// MODE 0: f16 C row-major        (Q projection)
// MODE 1: f32 C row-major        (final output projection)
// MODE 2: f16 C remapped into concatenated-K buffer rows 32.. per batch
// MODE 3: f16 C written transposed per-head into padded V^T buffer
template<int MODE>
__global__ __launch_bounds__(256)
void gemm_wmma(const _Float16* __restrict__ A, const _Float16* __restrict__ W,
               const float* __restrict__ bias, void* __restrict__ Cout,
               int K, int N)
{
  const int lane = threadIdx.x & 31;
  const int wid  = threadIdx.x >> 5;
  const int lh   = lane & 15;
  const int hi8  = (lane >> 4) << 3;   // A-frag K split 0/8
  const int hi16 = (lane >> 4) << 4;   // K split 0/16 (unused here; B-frag uses hi8)
  const int m0 = blockIdx.y * 128 + (wid >> 2) * 64;
  const int n0 = blockIdx.x * 256 + (wid & 3) * 64;

  v8f acc[4][4];
  #pragma unroll
  for (int i = 0; i < 4; ++i)
    #pragma unroll
    for (int j = 0; j < 4; ++j)
      acc[i][j] = (v8f){0.f,0.f,0.f,0.f,0.f,0.f,0.f,0.f};

  for (int k0 = 0; k0 < K; k0 += 32) {
    v16h af[4], bf[4];
    #pragma unroll
    for (int i = 0; i < 4; ++i) {
      const _Float16* ar = A + (size_t)(m0 + 16 * i + lh) * K + k0;
      af[i] = cmb(ld8(ar + hi8), ld8(ar + 16 + hi8));
    }
    #pragma unroll
    for (int j = 0; j < 4; ++j) {
      const _Float16* br = W + (size_t)(n0 + 16 * j + lh) * K + k0;
      bf[j] = cmb(ld8(br + hi8), ld8(br + 16 + hi8));
    }
    #pragma unroll
    for (int i = 0; i < 4; ++i)
      #pragma unroll
      for (int j = 0; j < 4; ++j)
        acc[i][j] = wmma16(af[i], bf[j], acc[i][j]);
  }

  #pragma unroll
  for (int i = 0; i < 4; ++i) {
    #pragma unroll
    for (int j = 0; j < 4; ++j) {
      const int n = n0 + 16 * j + lh;
      const float bv = bias[n];
      if (MODE == 3) {
        // lane holds 8 consecutive rows m (= consecutive s') at fixed column n
        const int mbase = m0 + 16 * i + hi8;
        const int bidx  = mbase >> 11;                 // batch
        const int srow  = NG + (mbase - (bidx << 11)); // s' in concat coords
        _Float16* dst = (_Float16*)Cout +
            (size_t)((bidx * NH + (n >> 6)) * HD + (n & 63)) * VT_LD + VT_OFF + srow;
        unsigned o0 = pkh2(acc[i][j][0] + bv, acc[i][j][1] + bv);
        unsigned o1 = pkh2(acc[i][j][2] + bv, acc[i][j][3] + bv);
        unsigned o2 = pkh2(acc[i][j][4] + bv, acc[i][j][5] + bv);
        unsigned o3 = pkh2(acc[i][j][6] + bv, acc[i][j][7] + bv);
        uint4 ov; ov.x = o0; ov.y = o1; ov.z = o2; ov.w = o3;
        *(uint4*)dst = ov;
      } else {
        #pragma unroll
        for (int r = 0; r < 8; ++r) {
          const int m = m0 + 16 * i + r + hi8;
          const float val = acc[i][j][r] + bv;
          if (MODE == 1) {
            ((float*)Cout)[(size_t)m * N + n] = val;
          } else if (MODE == 0) {
            ((_Float16*)Cout)[(size_t)m * N + n] = (_Float16)val;
          } else { // MODE == 2: Kcat row = b*2080 + 32 + s
            const int row = m + 32 * (m >> 11) + 32;
            ((_Float16*)Cout)[(size_t)row * N + n] = (_Float16)val;
          }
        }
      }
    }
  }
}

// ---------------- sparse attention ----------------
// One wave handles 16 query positions of one (b,h). 14 column-blocks of 16:
//   blk 0..1  : global tokens s'=0..31
//   blk 2..11 : local span sp in [t0-64, t0+96) (union mask local|strided)
//   blk 12..13: strided sp=64j, j=0..31 (masked out if inside local span)
__global__ __launch_bounds__(128)
void sparse_attn(const _Float16* __restrict__ Q, const _Float16* __restrict__ Kc,
                 const _Float16* __restrict__ Vt, const _Float16* __restrict__ Vs,
                 _Float16* __restrict__ ctx)
{
  const int lane = threadIdx.x & 31;
  const int wid  = threadIdx.x >> 5;
  const int lh   = lane & 15;
  const int hi8  = (lane >> 4) << 3;
  const int hi16 = (lane >> 4) << 4;
  const int b  = blockIdx.z, h = blockIdx.y;
  const int t0 = blockIdx.x * 64 + wid * 16;
  const int t  = t0 + lh;          // this lane's query position (N index)
  const int lbase = t0 - 64;       // local span start (sp coords)

  // Q as B-operand of S^T = K * Q^T : lane holds d hi8..hi8+7 and 16+hi8..16+hi8+7 per frag
  const _Float16* qr = Q + (size_t)(b * TT + t) * DM + h * HD;
  const v16h bq0 = cmb(ld8(qr + hi8),      ld8(qr + 16 + hi8));    // d 0..31
  const v16h bq1 = cmb(ld8(qr + 32 + hi8), ld8(qr + 48 + hi8));   // d 32..63

  // ---- stage 1: scores S^T (s' rows x t cols), 14 blocks ----
  v8f acc[14];
  #pragma unroll
  for (int blk = 0; blk < 14; ++blk) {
    int srow;
    if (blk < 2) {
      srow = blk * 16 + lh;
    } else if (blk < 12) {
      int sp = lbase + (blk - 2) * 16 + lh;
      sp = sp < 0 ? 0 : (sp > SS - 1 ? SS - 1 : sp);   // clamp addr; mask kills
      srow = NG + sp;
    } else {
      srow = NG + 64 * ((blk - 12) * 16 + lh);
    }
    const _Float16* ar = Kc + ((size_t)b * SPN + srow) * DM + h * HD;
    v16h a0 = cmb(ld8(ar + hi8),      ld8(ar + 16 + hi8));
    v16h a1 = cmb(ld8(ar + 32 + hi8), ld8(ar + 48 + hi8));
    v8f c = (v8f){0.f,0.f,0.f,0.f,0.f,0.f,0.f,0.f};
    c = wmma16(a0, bq0, c);
    c = wmma16(a1, bq1, c);
    acc[blk] = c;
  }

  // ---- mask + scale (1/sqrt(64)) ----
  #pragma unroll
  for (int blk = 0; blk < 14; ++blk) {
    #pragma unroll
    for (int r = 0; r < 8; ++r) {
      const int m = r + hi8;  // s' index within block
      bool ok;
      if (blk < 2) {
        ok = true;                                    // globals always attended
      } else if (blk < 12) {
        const int sp = lbase + (blk - 2) * 16 + m;
        ok = (sp >= 0) && (sp < SS) &&
             (((sp >= t - 64) && (sp < t + 64)) || ((sp & 63) == 0));
      } else {
        const int sp = 64 * ((blk - 12) * 16 + m);
        ok = (sp < lbase) || (sp >= lbase + 160);     // dedupe vs local span
      }
      acc[blk][r] = ok ? acc[blk][r] * 0.125f : -__builtin_inff();
    }
  }

  // ---- softmax over s' (per-lane regs + one half-swap shuffle) ----
  float mx = -__builtin_inff();
  #pragma unroll
  for (int blk = 0; blk < 14; ++blk)
    #pragma unroll
    for (int r = 0; r < 8; ++r)
      mx = fmaxf(mx, acc[blk][r]);
  mx = fmaxf(mx, __shfl_xor(mx, 16));

  float sum = 0.f;
  #pragma unroll
  for (int blk = 0; blk < 14; ++blk)
    #pragma unroll
    for (int r = 0; r < 8; ++r) {
      const float e = __expf(acc[blk][r] - mx);
      acc[blk][r] = e;
      sum += e;
    }
  sum += __shfl_xor(sum, 16);
  const float inv = 1.f / sum;

  // pack P to f16 pairs (accumulator layout == half of B-frag layout)
  unsigned pk[14][4];
  #pragma unroll
  for (int blk = 0; blk < 14; ++blk)
    #pragma unroll
    for (int jj = 0; jj < 4; ++jj)
      pk[blk][jj] = pkh2(acc[blk][2 * jj] * inv, acc[blk][2 * jj + 1] * inv);

  // ---- stage 2: O^T = V^T * P^T, 7 K-chunks of 32 s' ----
  v8f od[4];
  #pragma unroll
  for (int db = 0; db < 4; ++db)
    od[db] = (v8f){0.f,0.f,0.f,0.f,0.f,0.f,0.f,0.f};

  #pragma unroll
  for (int c = 0; c < 7; ++c) {
    // build B-fragment from two accumulator blocks with a half-swap
    unsigned fu[8];
    #pragma unroll
    for (int jj = 0; jj < 4; ++jj) {
      const unsigned a0 = pk[2 * c][jj], a1 = pk[2 * c + 1][jj];
      const unsigned s0 = __shfl_xor(a0, 16);
      const unsigned s1 = __shfl_xor(a1, 16);
      fu[jj]     = (lane < 16) ? a0 : s1;  // K 0..7  | 16..23
      fu[4 + jj] = (lane < 16) ? s0 : a1;  // K 8..15 | 24..31
    }
    union { unsigned u[8]; v16h v; } pb;
    #pragma unroll
    for (int jj = 0; jj < 8; ++jj) pb.u[jj] = fu[jj];

    #pragma unroll
    for (int db = 0; db < 4; ++db) {
      const int d = db * 16 + lh;
      const _Float16* vr;
      if (c == 6) {
        vr = Vs + (size_t)((b * NH + h) * HD + d) * NG;              // strided gather
      } else {
        const int sbase = (c == 0) ? 0 : (NG + lbase + (c - 1) * 32); // may be <0: guards are zero
        vr = Vt + (size_t)((b * NH + h) * HD + d) * VT_LD + VT_OFF + sbase;
      }
      v16h av = cmb(ld8(vr + hi16), ld8(vr + hi16 + 8));
      od[db] = wmma16(av, pb.v, od[db]);
    }
  }

  // ---- store ctx[t][h*64+d] : each lane holds 8 contiguous d ----
  #pragma unroll
  for (int db = 0; db < 4; ++db) {
    unsigned o0 = pkh2(od[db][0], od[db][1]);
    unsigned o1 = pkh2(od[db][2], od[db][3]);
    unsigned o2 = pkh2(od[db][4], od[db][5]);
    unsigned o3 = pkh2(od[db][6], od[db][7]);
    _Float16* cp = ctx + (size_t)(b * TT + t) * DM + h * HD + db * 16 + hi8;
    uint4 ov; ov.x = o0; ov.y = o1; ov.z = o2; ov.w = o3;
    *(uint4*)cp = ov;
  }
}

// ---------------- launch ----------------
extern "C" void kernel_launch(void* const* d_in, const int* in_sizes, int n_in,
                              void* d_out, int out_size, void* d_ws, size_t ws_size,
                              hipStream_t stream) {
  (void)in_sizes; (void)n_in; (void)out_size; (void)ws_size;
  const float* query = (const float*)d_in[0];
  const float* key   = (const float*)d_in[1];
  const float* value = (const float*)d_in[2];
  const float* Wq = (const float*)d_in[3];
  const float* bq = (const float*)d_in[4];
  const float* Wk = (const float*)d_in[5];
  const float* bk = (const float*)d_in[6];
  const float* Wv = (const float*)d_in[7];
  const float* bvv = (const float*)d_in[8];
  const float* Wo = (const float*)d_in[9];
  const float* bo = (const float*)d_in[10];
  const float* gt = (const float*)d_in[11];

  char* w = (char*)d_ws;
  auto carve = [&](size_t bytes) -> void* {
    void* p = (void*)w;
    w += (bytes + 255) & ~(size_t)255;
    return p;
  };

  _Float16* qH  = (_Float16*)carve((size_t)MROWS * DM * 2);
  _Float16* kH  = (_Float16*)carve((size_t)MROWS * DM * 2);
  _Float16* vH  = (_Float16*)carve((size_t)MROWS * DM * 2);
  _Float16* WqH = (_Float16*)carve((size_t)DM * DM * 2);
  _Float16* WkH = (_Float16*)carve((size_t)DM * DM * 2);
  _Float16* WvH = (_Float16*)carve((size_t)DM * DM * 2);
  _Float16* WoH = (_Float16*)carve((size_t)DM * DM * 2);
  _Float16* gH  = (_Float16*)carve((size_t)NG * DM * 2);
  _Float16* Qp  = (_Float16*)carve((size_t)MROWS * DM * 2);
  _Float16* Kc  = (_Float16*)carve((size_t)BB * SPN * DM * 2);
  _Float16* Vt  = (_Float16*)carve((size_t)BB * NH * HD * VT_LD * 2);
  _Float16* Vs  = (_Float16*)carve((size_t)BB * NH * HD * NG * 2);
  _Float16* ctx = (_Float16*)carve((size_t)MROWS * DM * 2);

  const int NELT = MROWS * DM;   // 4,194,304
  const int WN   = DM * DM;      // 1,048,576
  cast_f32_f16<<<(NELT + 255) / 256, 256, 0, stream>>>(query, qH, NELT);
  cast_f32_f16<<<(NELT + 255) / 256, 256, 0, stream>>>(key,   kH, NELT);
  cast_f32_f16<<<(NELT + 255) / 256, 256, 0, stream>>>(value, vH, NELT);
  cast_f32_f16<<<(WN + 255) / 256, 256, 0, stream>>>(Wq, WqH, WN);
  cast_f32_f16<<<(WN + 255) / 256, 256, 0, stream>>>(Wk, WkH, WN);
  cast_f32_f16<<<(WN + 255) / 256, 256, 0, stream>>>(Wv, WvH, WN);
  cast_f32_f16<<<(WN + 255) / 256, 256, 0, stream>>>(Wo, WoH, WN);
  cast_f32_f16<<<(NG * DM + 255) / 256, 256, 0, stream>>>(gt, gH, NG * DM);

  dim3 gg(DM / 256, MROWS / 128);  // (4, 32), 256 threads = 8 waves
  gemm_wmma<0><<<gg, 256, 0, stream>>>(qH, WqH, bq, Qp, DM, DM);
  gemm_wmma<2><<<gg, 256, 0, stream>>>(kH, WkH, bk, Kc, DM, DM);

  const int vtN = BB * NH * HD * VT_LD;
  zero_f16<<<(vtN + 255) / 256, 256, 0, stream>>>(Vt, vtN);
  gemm_wmma<3><<<gg, 256, 0, stream>>>(vH, WvH, bvv, Vt, DM, DM);

  const int gN = BB * NG * DM;
  kcat_globals<<<(gN + 255) / 256, 256, 0, stream>>>(gH, Kc);
  const int vgN = BB * NH * HD * NG;
  vt_globals<<<(vgN + 255) / 256, 256, 0, stream>>>(gH, Vt);
  vts_fill<<<(vgN + 255) / 256, 256, 0, stream>>>(Vt, Vs);

  sparse_attn<<<dim3(TT / 64, NH, BB), 128, 0, stream>>>(Qp, Kc, Vt, Vs, ctx);

  gemm_wmma<1><<<gg, 256, 0, stream>>>(ctx, WoH, bo, d_out, DM, DM);
}
